// MyRNN_6064493822085
// MI455X (gfx1250) — compile-verified, matrix operand order checked by reading, not run on hardware
//
#include <hip/hip_runtime.h>
#include <hip/hip_bf16.h>

// ---------------------------------------------------------------------------
// Elman RNN for MI455X (gfx1250, wave32, WMMA).
//   Phase 1: Z[t*B+b, u] = x[b,t,:] @ Wx + bias      (parallel bf16 WMMA GEMM)
//   Phase 2: h_t = tanh(Z_t + h_{t-1} @ Wh)          (persistent kernel,
//            512 sequential steps, device atomic barrier per step,
//            Wh held entirely in VGPRs, next-step Z prefetched)
// All matrix math uses v_wmma_f32_16x16x32_bf16 (fp32 accumulate).
// ---------------------------------------------------------------------------

#define B_DIM 128
#define T_DIM 512
#define D_DIM 128
#define U_DIM 512

typedef __attribute__((ext_vector_type(16))) __bf16 v16bf;
typedef __attribute__((ext_vector_type(8)))  float  v8f;

union FragU {
    v16bf v;
    uint4 q[2];
    __bf16 b[16];
};

// ---------------------------------------------------------------------------
// Workspace layout (bytes):
//   [0,       524288)  Wh packed bf16 B-fragments  (16 kc * 32 nt * 512)
//   [524288,  655360)  Wx packed bf16 B-fragments  ( 4 kc * 32 nt * 512)
//   [655360,  786432)  h ping buffer (bf16, row-major 128x512)
//   [786432,  917504)  h pong buffer
//   [917504,  917508)  grid barrier counter
// ---------------------------------------------------------------------------

__global__ void init_kernel(unsigned* __restrict__ h0_as_u32, unsigned* __restrict__ bar) {
    int i = blockIdx.x * blockDim.x + threadIdx.x;
    if (i < (B_DIM * U_DIM / 2)) h0_as_u32[i] = 0u;   // h_{-1} = 0
    if (i == 0) *bar = 0u;
}

// Pack a row-major fp32 [K, 512] weight into bf16 WMMA B-fragment order.
// Fragment (kc, nt) holds 32 lanes x 16 elements; element j of lane maps to
// K = kc*32 + ((j>>3)<<4) + ((lane>>4)<<3) + (j&7), N = nt*16 + (lane&15).
__global__ void pack_b_kernel(const float* __restrict__ src,
                              __bf16* __restrict__ dst, int KC) {
    int i = blockIdx.x * blockDim.x + threadIdx.x;
    if (i >= KC * 32 * 512) return;
    int j    = i & 15;
    int lane = (i >> 4) & 31;
    int nt   = (i >> 9) & 31;
    int kc   = i >> 14;
    int n = (nt << 4) + (lane & 15);
    int k = (kc << 5) + ((j >> 3) << 4) + ((lane >> 4) << 3) + (j & 7);
    dst[i] = (__bf16)src[k * U_DIM + n];
}

// Phase 1: Z = x @ Wx + b, written to d_out as [T*B, U].
// One wave per 16x16 output tile; K = 128 -> 4 WMMAs.
__global__ void __launch_bounds__(256) zgemm_kernel(
    const float* __restrict__ x, const __bf16* __restrict__ wxp,
    const float* __restrict__ bias, float* __restrict__ z) {
    int wave = (blockIdx.x << 3) + (threadIdx.x >> 5);
    int lane = threadIdx.x & 31;
    int nt = wave & 31;          // 0..31 column tiles
    int rt = wave >> 5;          // 0..4095 row tiles over T*B
    int mrow  = (rt << 4) + (lane & 15);   // global row r = t*128 + b
    int b     = mrow & 127;
    int t     = mrow >> 7;
    int kbase = (lane >> 4) << 3;
    int ncol  = (nt << 4) + (lane & 15);

    const float* xrow = x + ((size_t)b * T_DIM + t) * D_DIM;

    float bv = bias[ncol];
    v8f c;
#pragma unroll
    for (int v = 0; v < 8; ++v) c[v] = bv;

#pragma unroll
    for (int kc = 0; kc < 4; ++kc) {
        const float* pa = xrow + (kc << 5) + kbase;
        float4 f0 = *(const float4*)(pa);
        float4 f1 = *(const float4*)(pa + 4);
        float4 f2 = *(const float4*)(pa + 16);
        float4 f3 = *(const float4*)(pa + 20);
        FragU a;
        a.b[0]  = (__bf16)f0.x; a.b[1]  = (__bf16)f0.y; a.b[2]  = (__bf16)f0.z; a.b[3]  = (__bf16)f0.w;
        a.b[4]  = (__bf16)f1.x; a.b[5]  = (__bf16)f1.y; a.b[6]  = (__bf16)f1.z; a.b[7]  = (__bf16)f1.w;
        a.b[8]  = (__bf16)f2.x; a.b[9]  = (__bf16)f2.y; a.b[10] = (__bf16)f2.z; a.b[11] = (__bf16)f2.w;
        a.b[12] = (__bf16)f3.x; a.b[13] = (__bf16)f3.y; a.b[14] = (__bf16)f3.z; a.b[15] = (__bf16)f3.w;

        FragU bf;
        const uint4* bp = (const uint4*)(wxp + (((kc << 5) + nt) << 9) + (lane << 4));
        bf.q[0] = bp[0]; bf.q[1] = bp[1];

        c = __builtin_amdgcn_wmma_f32_16x16x32_bf16(false, a.v, false, bf.v,
                                                    (short)0, c, false, false);
    }

#pragma unroll
    for (int v = 0; v < 8; ++v) {
        int row = (rt << 4) + v + ((lane >> 4) << 3);
        z[(size_t)row * U_DIM + ncol] = c[v];
    }
}

// Phase 2: persistent recurrence. 32 blocks x 256 threads = 256 waves, one
// 16x16 tile of the [128,512] h matrix per wave. The wave's Wh column tile
// (16 K-fragments, 128 VGPRs) is register-resident for all 512 steps.
// Per step: C <- Z_t tile, 16x v_wmma over K=512, tanh, write fp32 h_t to
// d_out and bf16 h_t to the ping-pong buffer, then device-wide barrier.
__global__ void __launch_bounds__(256) rnn_kernel(
    const __bf16* __restrict__ whp, float* __restrict__ z,
    __bf16* __restrict__ hb0, __bf16* __restrict__ hb1,
    unsigned* __restrict__ bar) {
    int wave = (blockIdx.x << 3) + (threadIdx.x >> 5);  // 0..255
    int lane = threadIdx.x & 31;
    int nt = wave & 31;                 // 0..31 column tiles of U
    int mt = wave >> 5;                 // 0..7  row tiles of B
    int mrow  = (mt << 4) + (lane & 15);
    int kbase = (lane >> 4) << 3;
    int ncol  = (nt << 4) + (lane & 15);

    // Hoist the wave's entire Wh column tile into registers (invariant in t).
    FragU bfr[16];
#pragma unroll
    for (int kc = 0; kc < 16; ++kc) {
        const uint4* bp = (const uint4*)(whp + (((kc << 5) + nt) << 9) + (lane << 4));
        bfr[kc].q[0] = bp[0];
        bfr[kc].q[1] = bp[1];
    }

    __bf16* hbufs[2] = {hb0, hb1};

    for (int t = 0; t < T_DIM; ++t) {
        const __bf16* hin  = hbufs[t & 1];
        __bf16*       hout = hbufs[(t + 1) & 1];
        float* zt = z + (size_t)t * (B_DIM * U_DIM);

        // Prefetch next step's Z tile (one 64B row-chunk per lane 0..15).
        if (t + 1 < T_DIM) {
            const float* ztn = zt + (B_DIM * U_DIM);
            __builtin_prefetch(ztn + ((mt << 4) + (lane & 15)) * U_DIM + (nt << 4), 0, 3);
        }

        v8f c;
#pragma unroll
        for (int v = 0; v < 8; ++v) {
            int row = (mt << 4) + v + ((lane >> 4) << 3);
            c[v] = zt[row * U_DIM + ncol];
        }

        const __bf16* arow = hin + mrow * U_DIM;
#pragma unroll
        for (int kc = 0; kc < 16; ++kc) {
            FragU a;
            a.q[0] = *(const uint4*)(arow + (kc << 5) + kbase);
            a.q[1] = *(const uint4*)(arow + (kc << 5) + kbase + 16);
            c = __builtin_amdgcn_wmma_f32_16x16x32_bf16(false, a.v, false, bfr[kc].v,
                                                        (short)0, c, false, false);
        }

#pragma unroll
        for (int v = 0; v < 8; ++v) {
            int row = (mt << 4) + v + ((lane >> 4) << 3);
            float hv = tanhf(c[v]);
            zt[row * U_DIM + ncol]   = hv;           // output h_t (fp32)
            hout[row * U_DIM + ncol] = (__bf16)hv;   // recurrent state (bf16)
        }

        // device-wide barrier: monotonic counter, target = (t+1)*gridDim.x
        __syncthreads();
        if (threadIdx.x == 0) {
            __threadfence();
            __hip_atomic_fetch_add(bar, 1u, __ATOMIC_ACQ_REL, __HIP_MEMORY_SCOPE_AGENT);
            unsigned target = (unsigned)(t + 1) * gridDim.x;
            while (__hip_atomic_load(bar, __ATOMIC_ACQUIRE, __HIP_MEMORY_SCOPE_AGENT) < target)
                __builtin_amdgcn_s_sleep(1);
        }
        __syncthreads();
        __threadfence();
    }
}

extern "C" void kernel_launch(void* const* d_in, const int* in_sizes, int n_in,
                              void* d_out, int out_size, void* d_ws, size_t ws_size,
                              hipStream_t stream) {
    const float* x    = (const float*)d_in[0];  // [B,T,D]
    const float* Wx   = (const float*)d_in[1];  // [D,U]
    const float* Wh   = (const float*)d_in[2];  // [U,U]
    const float* bias = (const float*)d_in[3];  // [U]
    float* out = (float*)d_out;                 // [T,B,U]

    char* ws = (char*)d_ws;                     // needs ~918 KB
    __bf16*   whp = (__bf16*)(ws);              // 512 KB
    __bf16*   wxp = (__bf16*)(ws + 524288);     // 128 KB
    __bf16*   hb0 = (__bf16*)(ws + 655360);     // 128 KB
    __bf16*   hb1 = (__bf16*)(ws + 786432);     // 128 KB
    unsigned* bar = (unsigned*)(ws + 917504);

    hipLaunchKernelGGL(init_kernel, dim3(128), dim3(256), 0, stream,
                       (unsigned*)hb0, bar);
    hipLaunchKernelGGL(pack_b_kernel, dim3(1024), dim3(256), 0, stream,
                       Wh, whp, 16);
    hipLaunchKernelGGL(pack_b_kernel, dim3(256), dim3(256), 0, stream,
                       Wx, wxp, 4);
    hipLaunchKernelGGL(zgemm_kernel, dim3(16384), dim3(256), 0, stream,
                       x, wxp, bias, out);
    hipLaunchKernelGGL(rnn_kernel, dim3(32), dim3(256), 0, stream,
                       whp, out, hb0, hb1, bar);
}
